// TransformNet_53781580480528
// MI455X (gfx1250) — matrix operand order repaired and checked by reading, compiled, not run on hardware
//
#include <hip/hip_runtime.h>
#include <hip/hip_bf16.h>
#include <math.h>

// ---------------------------------------------------------------------------
// Types for CDNA5 WMMA (wave32, 16x16x32 bf16 -> f32) and TDM descriptors
// ---------------------------------------------------------------------------
typedef __attribute__((ext_vector_type(16))) __bf16 v16bf;
typedef __attribute__((ext_vector_type(8)))  __bf16 v8bf;
typedef __attribute__((ext_vector_type(8)))  float  v8f;
typedef __attribute__((ext_vector_type(4)))  unsigned int v4u;
typedef __attribute__((ext_vector_type(8)))  int v8i;
typedef __attribute__((ext_vector_type(4)))  int v4i;

#define BN_RSQ   0.9999950000374998f   // 1/sqrt(1+1e-5)
#define RSQRT_C  0.08838834764831845f  // 1/sqrt(128)
#define LDSROW   136                   // B-tile LDS row stride in halves (272B: bank-skewed, 16B aligned)

__device__ __forceinline__ unsigned short f2bf(float x) {
  unsigned u = __float_as_uint(x);
  u += 0x7fffu + ((u >> 16) & 1u);          // round-to-nearest-even
  return (unsigned short)(u >> 16);
}
__device__ __forceinline__ int f2ord(float f) {   // monotonic float->int map
  int i = __float_as_int(f);
  return i >= 0 ? i : (i ^ 0x7fffffff);
}
__device__ __forceinline__ float ord2f(int i) {
  return __int_as_float(i >= 0 ? i : (i ^ 0x7fffffff));
}

// ---------------------------------------------------------------------------
// Utility fills / converts
// ---------------------------------------------------------------------------
__global__ void k_fill_u32(unsigned* __restrict__ p, unsigned val, int n) {
  int i = blockIdx.x * blockDim.x + threadIdx.x;
  if (i < n) p[i] = val;
}

__global__ void k_cvt_bf16(const float* __restrict__ x, unsigned short* __restrict__ y, int n) {
  int i = blockIdx.x * blockDim.x + threadIdx.x;
  if (i < n) y[i] = f2bf(x[i]);
}

// W[K][M] (f32, row-major) -> Wt[M][K] (bf16)  (K fixed 128)
__global__ void k_wt_bf16(const float* __restrict__ W, unsigned short* __restrict__ Wt, int M) {
  int id = blockIdx.x * blockDim.x + threadIdx.x;
  if (id >= M * 128) return;
  int m = id >> 7, kk = id & 127;
  Wt[id] = f2bf(W[kk * M + m]);
}

// ---------------------------------------------------------------------------
// BN on 5-wide input
// ---------------------------------------------------------------------------
__global__ void k_bn5(const float* __restrict__ x, const float* __restrict__ g,
                      const float* __restrict__ b, float* __restrict__ y, int n5) {
  int i = blockIdx.x * blockDim.x + threadIdx.x;
  if (i >= n5) return;
  int c = i % 5;
  y[i] = x[i] * (g[c] * BN_RSQ) + b[c];
}

// ---------------------------------------------------------------------------
// conv1 projections (cin=5): q,k,v (256 wide) + skip (128 wide)
// ---------------------------------------------------------------------------
__global__ void k_conv1(const float* __restrict__ hb,
                        const float* __restrict__ Wq, const float* __restrict__ bq,
                        const float* __restrict__ Wk, const float* __restrict__ bk,
                        const float* __restrict__ Wv, const float* __restrict__ bv,
                        const float* __restrict__ Ws, const float* __restrict__ bs,
                        float* __restrict__ q, float* __restrict__ k,
                        float* __restrict__ v, float* __restrict__ skip, int N) {
  int n = blockIdx.x;
  int j = threadIdx.x;                       // 0..255
  if (n >= N) return;
  float x0 = hb[n * 5 + 0], x1 = hb[n * 5 + 1], x2 = hb[n * 5 + 2];
  float x3 = hb[n * 5 + 3], x4 = hb[n * 5 + 4];
  float aq = bq[j] + x0 * Wq[j] + x1 * Wq[256 + j] + x2 * Wq[512 + j] + x3 * Wq[768 + j] + x4 * Wq[1024 + j];
  float ak = bk[j] + x0 * Wk[j] + x1 * Wk[256 + j] + x2 * Wk[512 + j] + x3 * Wk[768 + j] + x4 * Wk[1024 + j];
  float av = bv[j] + x0 * Wv[j] + x1 * Wv[256 + j] + x2 * Wv[512 + j] + x3 * Wv[768 + j] + x4 * Wv[1024 + j];
  size_t o = (size_t)n * 256 + j;
  q[o] = aq; k[o] = ak; v[o] = av;
  if (j < 128) {
    float as = bs[j] + x0 * Ws[j] + x1 * Ws[128 + j] + x2 * Ws[256 + j] + x3 * Ws[384 + j] + x4 * Ws[512 + j];
    skip[(size_t)n * 128 + j] = as;
  }
}

// ---------------------------------------------------------------------------
// WMMA GEMM with TDM-staged B tile in LDS.
//   C[N x M] = A_bf16[N x 128] * Wt_bf16[M x 128]^T + bias
// Block: 256 threads = 8 waves; block computes 128 rows x 64 cols.
// B tile (64 cols x 128 K, bf16) is DMA'd into LDS once per block by the
// Tensor Data Mover with pad_enable (16B pad per 256B row -> 272B stride, so
// lanes 0..15 (cols) hit distinct LDS banks). grid = (Npad/128, M/64).
// ---------------------------------------------------------------------------
__global__ __launch_bounds__(256) void k_wmma_gemm(const unsigned short* __restrict__ A,
                                                   const unsigned short* __restrict__ Wt,
                                                   const float* __restrict__ bias,
                                                   float* __restrict__ C, int N, int M) {
  __shared__ unsigned short Bt[64 * LDSROW];   // 17408 B

  const int tid  = threadIdx.x;
  const int wv   = tid >> 5;                   // wave 0..7
  const int lane = tid & 31;
  const int m16  = lane & 15;
  const int agrp = lane >> 4;                  // lane half: 0/1
  const int row0 = (blockIdx.x * 8 + wv) * 16;
  const int col0 = blockIdx.y * 64;

#if __has_builtin(__builtin_amdgcn_tensor_load_to_lds)
  if (wv == 0) {
    // ---- Tensor DMA descriptor (ISA cdna5 8.3-8.6), 2-D tile 128 x 64 ----
    unsigned long long ga = (unsigned long long)(uintptr_t)(Wt + (size_t)col0 * 128);
    unsigned ldsa = (unsigned)(uintptr_t)(void*)&Bt[0];   // flat low 32 bits == LDS byte addr
    v4u g0;
    g0[0] = 1u;                                            // count=1, user descriptor
    g0[1] = ldsa;                                          // lds_addr
    g0[2] = (unsigned)ga;                                  // global_addr[31:0]
    g0[3] = (unsigned)((ga >> 32) & 0x01FFFFFFull) | 0x80000000u; // addr[56:32] | type=2
    v8i g1;
    g1[0] = (int)((1u << 16)        // data_size = 2 bytes
                | (1u << 20)        // pad_enable
                | (5u << 22)        // pad_interval: 64 DWORDs (256B)
                | (3u << 25));      // pad_amount: 4 DWORDs (16B)
    g1[1] = (int)(128u << 16);      // tensor_dim0 = 128 (halves per row)
    g1[2] = (int)((unsigned)M << 16);   // tensor_dim0 hi=0 | tensor_dim1 lo = M
    g1[3] = (int)(128u << 16);      // tensor_dim1 hi=0 | tile_dim0 = 128
    g1[4] = 64;                     // tile_dim1 = 64 rows, tile_dim2 = 0
    g1[5] = 128;                    // tensor_dim0_stride = 128
    g1[6] = 0;                      // stride0 hi | tensor_dim1_stride lo
    g1[7] = 0;
    v4i gz = {0, 0, 0, 0};
#if __clang_major__ >= 23
    v8i gz8 = {0, 0, 0, 0, 0, 0, 0, 0};
    __builtin_amdgcn_tensor_load_to_lds(g0, g1, gz, gz, gz8, 0);
#else
    __builtin_amdgcn_tensor_load_to_lds(g0, g1, gz, gz, 0);
#endif
#if __has_builtin(__builtin_amdgcn_s_wait_tensorcnt)
    __builtin_amdgcn_s_wait_tensorcnt(0);
#else
    asm volatile("s_wait_tensorcnt 0x0" ::: "memory");
#endif
  }
#else
  // Cooperative fallback: copy 64 rows x 256B with 272B LDS stride.
  for (int idx = tid; idx < 64 * 8; idx += 256) {
    int r = idx >> 3, seg = idx & 7;
    const uint4* s = reinterpret_cast<const uint4*>(Wt + (size_t)(col0 + r) * 128 + seg * 8);
    *reinterpret_cast<uint4*>(&Bt[r * LDSROW + seg * 8]) = *s;
  }
#endif
  __syncthreads();

  v8f acc[4] = {};
  const unsigned short* arow = A + (size_t)(row0 + m16) * 128;

#pragma unroll
  for (int kb = 0; kb < 128; kb += 32) {
    // A fragment (ISA 16-bit A 16x32 layout): halves 0..7 = K kb+agrp*8..+7,
    // halves 8..15 = K kb+agrp*8+16..+23
    v16bf af;
    {
      const v8bf* ap = reinterpret_cast<const v8bf*>(arow + kb + agrp * 8);
      v8bf lo = ap[0], hi = ap[2];
#pragma unroll
      for (int i = 0; i < 8; ++i) { af[i] = lo[i]; af[i + 8] = hi[i]; }
    }
#pragma unroll
    for (int t = 0; t < 4; ++t) {
      // B fragment from LDS: lane holds column t*16+m16, K = kb+agrp*8..+7, kb+agrp*8+16..+23
      const unsigned short* brow = &Bt[(t * 16 + m16) * LDSROW + kb + agrp * 8];
      const v8bf* bp = reinterpret_cast<const v8bf*>(brow);
      v8bf lo = bp[0], hi = bp[2];
      v16bf bfr;
#pragma unroll
      for (int i = 0; i < 8; ++i) { bfr[i] = lo[i]; bfr[i + 8] = hi[i]; }
      acc[t] = __builtin_amdgcn_wmma_f32_16x16x32_bf16(
          /*neg_a=*/false, af, /*neg_b=*/false, bfr,
          /*c_mod=*/(short)0, acc[t], /*reuse_a=*/false, /*reuse_b=*/false);
    }
  }

  // C/D layout: VGPR r -> row (row0 + agrp*8 + r), col = col0 + t*16 + m16
#pragma unroll
  for (int t = 0; t < 4; ++t) {
    int col = col0 + t * 16 + m16;
    float bia = bias[col];
#pragma unroll
    for (int r = 0; r < 8; ++r) {
      int row = row0 + agrp * 8 + r;
      if (row < N) C[(size_t)row * M + col] = acc[t][r] + bia;
    }
  }
}

// ---------------------------------------------------------------------------
// Edge kernels: one wave per (edge, head) pair; lane covers 4 channels
// ---------------------------------------------------------------------------
__global__ void k_edge_alpha(const float* __restrict__ q, const float* __restrict__ k,
                             const float* __restrict__ We, const float* __restrict__ ew,
                             const int* __restrict__ src, const int* __restrict__ dst,
                             float* __restrict__ alpha, int* __restrict__ mmax, int npairs) {
  int wid = blockIdx.x * (blockDim.x >> 5) + (threadIdx.x >> 5);
  if (wid >= npairs) return;
  int lane = threadIdx.x & 31;
  int e = wid >> 1, h = wid & 1;
  int s = src[e], d = dst[e];
  float w = ew[e];
  const float4* qv = reinterpret_cast<const float4*>(q + (size_t)d * 256 + h * 128);
  const float4* kv = reinterpret_cast<const float4*>(k + (size_t)s * 256 + h * 128);
  const float4* ev = reinterpret_cast<const float4*>(We + h * 128);
  float4 qq = qv[lane], kk = kv[lane], ee = ev[lane];
  float a = qq.x * fmaf(w, ee.x, kk.x) + qq.y * fmaf(w, ee.y, kk.y)
          + qq.z * fmaf(w, ee.z, kk.z) + qq.w * fmaf(w, ee.w, kk.w);
#pragma unroll
  for (int off = 16; off > 0; off >>= 1) a += __shfl_xor(a, off, 32);
  if (lane == 0) {
    a *= RSQRT_C;
    alpha[wid] = a;
    atomicMax(mmax + (size_t)d * 2 + h, f2ord(a));
  }
}

// Fused exp + sum + unnormalized message scatter (normalization in finalize).
__global__ void k_edge_msg(const float* __restrict__ alpha, const float* __restrict__ v,
                           const float* __restrict__ We, const float* __restrict__ ew,
                           const int* __restrict__ src, const int* __restrict__ dst,
                           const int* __restrict__ mmax, float* __restrict__ ssum,
                           float* __restrict__ agg, int npairs) {
  int wid = blockIdx.x * (blockDim.x >> 5) + (threadIdx.x >> 5);
  if (wid >= npairs) return;
  int lane = threadIdx.x & 31;
  int e = wid >> 1, h = wid & 1;
  int s = src[e], d = dst[e];
  float w = ew[e];
  float m = ord2f(mmax[(size_t)d * 2 + h]);
  float ex = __expf(alpha[wid] - m);
  if (lane == 0) atomicAdd(ssum + (size_t)d * 2 + h, ex);
  const float4* vv = reinterpret_cast<const float4*>(v + (size_t)s * 256 + h * 128);
  const float4* ev = reinterpret_cast<const float4*>(We + h * 128);
  float4 vx = vv[lane], ee = ev[lane];
  float* ag = agg + (size_t)d * 256 + h * 128 + lane * 4;
  atomicAdd(ag + 0, fmaf(w, ee.x, vx.x) * ex);
  atomicAdd(ag + 1, fmaf(w, ee.y, vx.y) * ex);
  atomicAdd(ag + 2, fmaf(w, ee.z, vx.z) * ex);
  atomicAdd(ag + 3, fmaf(w, ee.w, vx.w) * ex);
}

// ---------------------------------------------------------------------------
// Epilogues
// ---------------------------------------------------------------------------
// out = bn(relu(mean_heads(agg/s) + skip)) [+ res]
__global__ void k_finalize(const float* __restrict__ agg, const float* __restrict__ ssum,
                           const float* __restrict__ skip,
                           const float* __restrict__ g, const float* __restrict__ b,
                           const float* __restrict__ res, float* __restrict__ out, int n128) {
  int i = blockIdx.x * blockDim.x + threadIdx.x;
  if (i >= n128) return;
  int n = i >> 7, c = i & 127;
  float s0 = ssum[(size_t)n * 2 + 0] + 1e-16f;
  float s1 = ssum[(size_t)n * 2 + 1] + 1e-16f;
  float t = 0.5f * (agg[(size_t)n * 256 + c] / s0 + agg[(size_t)n * 256 + 128 + c] / s1) + skip[i];
  t = fmaxf(t, 0.0f);
  t = t * (g[c] * BN_RSQ) + b[c];
  if (res) t += res[i];
  out[i] = t;
}

// out = bn(relu(x)) + res
__global__ void k_relu_bn_res(const float* __restrict__ x, const float* __restrict__ g,
                              const float* __restrict__ b, const float* __restrict__ res,
                              float* __restrict__ out, int n128) {
  int i = blockIdx.x * blockDim.x + threadIdx.x;
  if (i >= n128) return;
  int c = i & 127;
  out[i] = fmaxf(x[i], 0.0f) * (g[c] * BN_RSQ) + b[c] + res[i];
}

// final fc2: [N x 128] @ [128 x 5] + bias
__global__ void k_fc2(const float* __restrict__ h, const float* __restrict__ W,
                      const float* __restrict__ b, float* __restrict__ out, int N) {
  int n = blockIdx.x * blockDim.x + threadIdx.x;
  if (n >= N) return;
  float a0 = b[0], a1 = b[1], a2 = b[2], a3 = b[3], a4 = b[4];
  const float* hr = h + (size_t)n * 128;
#pragma unroll 4
  for (int c = 0; c < 128; ++c) {
    float hv = hr[c];
    const float* wr = W + c * 5;
    a0 += hv * wr[0]; a1 += hv * wr[1]; a2 += hv * wr[2]; a3 += hv * wr[3]; a4 += hv * wr[4];
  }
  float* o = out + (size_t)n * 5;
  o[0] = a0; o[1] = a1; o[2] = a2; o[3] = a3; o[4] = a4;
}

// ---------------------------------------------------------------------------
// Host orchestration
// ---------------------------------------------------------------------------
enum {
  IN_H = 0, IN_EIDX = 1, IN_EW = 2,
  C1_WQ = 3, C1_BQ, C1_WK, C1_BK, C1_WV, C1_BV, C1_WE, C1_WS, C1_BS,
  C2_WQ = 12, C2_BQ, C2_WK, C2_BK, C2_WV, C2_BV, C2_WE, C2_WS, C2_BS,
  C3_WQ = 21, C3_BQ, C3_WK, C3_BK, C3_WV, C3_BV, C3_WE, C3_WS, C3_BS,
  P_WFC1 = 30, P_BFC1, P_WFC2, P_BFC2,
  P_G0 = 34, P_B0, P_G1, P_B1, P_G2, P_B2, P_G3, P_B3, P_GF, P_BF
};

extern "C" void kernel_launch(void* const* d_in, const int* in_sizes, int n_in,
                              void* d_out, int out_size, void* d_ws, size_t ws_size,
                              hipStream_t stream) {
  const int N = in_sizes[IN_H] / 5;
  const int E = in_sizes[IN_EW];
  const int Npad = (N + 127) & ~127;       // GEMM row padding (128-row blocks)

  const float* h_in = (const float*)d_in[IN_H];
  const int*   eidx = (const int*)d_in[IN_EIDX];
  const int*   src  = eidx;
  const int*   dst  = eidx + E;
  const float* ew   = (const float*)d_in[IN_EW];
  auto F = [&](int i) { return (const float*)d_in[i]; };

  // ---- workspace carve (256B aligned) ----
  char* wp = (char*)d_ws;
  auto alloc = [&](size_t bytes) { void* r = wp; wp += (bytes + 255) & ~(size_t)255; return r; };
  float* hb5  = (float*)alloc((size_t)N * 5 * 4);
  float* hA   = (float*)alloc((size_t)N * 128 * 4);
  float* hB   = (float*)alloc((size_t)N * 128 * 4);
  unsigned short* hbf = (unsigned short*)alloc((size_t)Npad * 128 * 2);
  float* qb   = (float*)alloc((size_t)N * 256 * 4);
  float* kb   = (float*)alloc((size_t)N * 256 * 4);
  float* vb   = (float*)alloc((size_t)N * 256 * 4);
  float* skip = (float*)alloc((size_t)N * 128 * 4);
  float* agg  = (float*)alloc((size_t)N * 256 * 4);
  float* alp  = (float*)alloc((size_t)E * 2 * 4);
  int*   mmax = (int*)alloc((size_t)N * 2 * 4);
  float* ssum = (float*)alloc((size_t)N * 2 * 4);
  unsigned short* wtq2 = (unsigned short*)alloc(256 * 128 * 2);
  unsigned short* wtk2 = (unsigned short*)alloc(256 * 128 * 2);
  unsigned short* wtv2 = (unsigned short*)alloc(256 * 128 * 2);
  unsigned short* wts2 = (unsigned short*)alloc(128 * 128 * 2);
  unsigned short* wtq3 = (unsigned short*)alloc(256 * 128 * 2);
  unsigned short* wtk3 = (unsigned short*)alloc(256 * 128 * 2);
  unsigned short* wtv3 = (unsigned short*)alloc(256 * 128 * 2);
  unsigned short* wts3 = (unsigned short*)alloc(128 * 128 * 2);
  unsigned short* wtf1 = (unsigned short*)alloc(128 * 128 * 2);

  auto fill = [&](void* p, unsigned val, int n) {
    k_fill_u32<<<(n + 255) / 256, 256, 0, stream>>>((unsigned*)p, val, n);
  };
  auto prep = [&](const float* W, unsigned short* Wt, int M) {
    k_wt_bf16<<<(M * 128 + 255) / 256, 256, 0, stream>>>(W, Wt, M);
  };

  // ---- weight conversion (bf16, transposed to [M][K]) ----
  prep(F(C2_WQ), wtq2, 256); prep(F(C2_WK), wtk2, 256); prep(F(C2_WV), wtv2, 256);
  prep(F(C2_WS), wts2, 128);
  prep(F(C3_WQ), wtq3, 256); prep(F(C3_WK), wtk3, 256); prep(F(C3_WV), wtv3, 256);
  prep(F(C3_WS), wts3, 128);
  prep(F(P_WFC1), wtf1, 128);

  const int npairs = E * 2;
  const int gridw = (npairs + 7) / 8;    // 8 waves per 256-thread block

  auto run_edges = [&](const float* q, const float* k, const float* v, const float* We) {
    fill(mmax, 0x807FFFFFu, N * 2);      // f2ord(-inf)
    fill(ssum, 0u, N * 2);
    fill(agg, 0u, N * 256);
    k_edge_alpha<<<gridw, 256, 0, stream>>>(q, k, We, ew, src, dst, alp, mmax, npairs);
    k_edge_msg<<<gridw, 256, 0, stream>>>(alp, v, We, ew, src, dst, mmax, ssum, agg, npairs);
  };

  auto gemm = [&](const unsigned short* Wt, const float* bias, float* out, int M) {
    k_wmma_gemm<<<dim3(Npad / 128, M / 64), 256, 0, stream>>>(hbf, Wt, bias, out, N, M);
  };

  auto run_conv = [&](const float* xin,
                      const unsigned short* wq, const float* bq,
                      const unsigned short* wk, const float* bk2,
                      const unsigned short* wv, const float* bv,
                      const unsigned short* ws, const float* bs,
                      const float* We, const float* g, const float* bbn,
                      const float* res, float* out) {
    k_cvt_bf16<<<(N * 128 + 255) / 256, 256, 0, stream>>>(xin, hbf, N * 128);
    gemm(wq, bq, qb, 256);
    gemm(wk, bk2, kb, 256);
    gemm(wv, bv, vb, 256);
    gemm(ws, bs, skip, 128);
    run_edges(qb, kb, vb, We);
    k_finalize<<<(N * 128 + 255) / 256, 256, 0, stream>>>(agg, ssum, skip, g, bbn, res, out, N * 128);
  };

  // ---- layer 0: BN(5) + conv1 (cin=5, scalar projections) ----
  k_bn5<<<(N * 5 + 255) / 256, 256, 0, stream>>>(h_in, F(P_G0), F(P_B0), hb5, N * 5);
  k_conv1<<<N, 256, 0, stream>>>(hb5, F(C1_WQ), F(C1_BQ), F(C1_WK), F(C1_BK),
                                 F(C1_WV), F(C1_BV), F(C1_WS), F(C1_BS),
                                 qb, kb, vb, skip, N);
  run_edges(qb, kb, vb, F(C1_WE));
  k_finalize<<<(N * 128 + 255) / 256, 256, 0, stream>>>(agg, ssum, skip, F(P_G1), F(P_B1),
                                                        nullptr, hA, N * 128);

  // ---- conv2: in=hA, res=hA, out=hB ----
  run_conv(hA, wtq2, F(C2_BQ), wtk2, F(C2_BK), wtv2, F(C2_BV), wts2, F(C2_BS),
           F(C2_WE), F(P_G2), F(P_B2), hA, hB);

  // ---- conv3: in=hB, res=hB, out=hA ----
  run_conv(hB, wtq3, F(C3_BQ), wtk3, F(C3_BK), wtv3, F(C3_BV), wts3, F(C3_BS),
           F(C3_WE), F(P_G3), F(P_B3), hB, hA);

  // ---- fc1 + BN + residual: in=hA -> hB ----
  k_cvt_bf16<<<(N * 128 + 255) / 256, 256, 0, stream>>>(hA, hbf, N * 128);
  gemm(wtf1, F(P_BFC1), skip /*tmp*/, 128);
  k_relu_bn_res<<<(N * 128 + 255) / 256, 256, 0, stream>>>(skip, F(P_GF), F(P_BF),
                                                           hA, hB, N * 128);

  // ---- fc2 -> d_out ----
  k_fc2<<<(N + 127) / 128, 128, 0, stream>>>(hB, F(P_WFC2), F(P_BFC2), (float*)d_out, N);
}
